// VectorQuantizer_38465727102993
// MI455X (gfx1250) — matrix operand order repaired and checked by reading, not compiled, not run on hardware
//
#include <hip/hip_runtime.h>
#include <hip/hip_bf16.h>

typedef __attribute__((ext_vector_type(16))) _Float16 v16h;
typedef __attribute__((ext_vector_type(8)))  float    v8f;

#define KCODES 512
#define DIM 64
#define NROWS 65536            // 64 * 32 * 32
#define ROWS_PER_WAVE 32
#define ROWS_PER_BLOCK 256     // 8 waves * 32 rows

#define WMMA_F16(A, B, C) \
    __builtin_amdgcn_wmma_f32_16x16x32_f16(false, (A), false, (B), (short)0, (C), false, false)

// ---------------------------------------------------------------------------
// Prep: split codebook rows into f16 hi/lo parts and compute ||e_k||^2 (fp32).
// ---------------------------------------------------------------------------
__global__ __launch_bounds__(256) void vq_prep(const float* __restrict__ emb,
                                               _Float16* __restrict__ ehi,
                                               _Float16* __restrict__ elo,
                                               float* __restrict__ cnorm) {
    int k = blockIdx.x * blockDim.x + threadIdx.x;
    if (k >= KCODES) return;
    const float* row = emb + k * DIM;
    float cn = 0.f;
    #pragma unroll 8
    for (int d = 0; d < DIM; ++d) {
        float v = row[d];
        _Float16 hi = (_Float16)v;
        _Float16 lo = (_Float16)(v - (float)hi);
        ehi[k * DIM + d] = hi;
        elo[k * DIM + d] = lo;
        cn = fmaf(v, v, cn);
    }
    cnorm[k] = cn;
}

// Load one 16-row x 64-dim A tile from [B,D,H,W] latents, split fp32 -> f16 hi/lo.
// A 16x32 in this code's k order (B matches): lane m = lane&15 holds K = hl*16+0..7 (v0-3), hl*16+8+0..7 (v4-7).
__device__ inline void load_split_A(const float* __restrict__ latb, int hl,
                                    v16h aH[2], v16h aL[2]) {
    #pragma unroll
    for (int s = 0; s < 2; ++s) {
        v16h ah, al;
        const int dbase = s * 32 + hl * 16;
        #pragma unroll
        for (int j = 0; j < 8; ++j) {
            float x1 = latb[(dbase + j) * 1024];
            float x2 = latb[(dbase + 8 + j) * 1024];
            _Float16 h1 = (_Float16)x1;
            _Float16 h2 = (_Float16)x2;
            ah[j]     = h1;
            ah[8 + j] = h2;
            al[j]     = (_Float16)(x1 - (float)h1);
            al[8 + j] = (_Float16)(x2 - (float)h2);
        }
        aH[s] = ah; aL[s] = al;
    }
}

// ---------------------------------------------------------------------------
// Main: per wave, 32 rows x 512 codes via compensated f16 WMMA with two
// interleaved accumulator chains sharing every B load; argmin; then
// block-wide exact fp32 gather/write + deterministic loss partial.
// ---------------------------------------------------------------------------
__global__ __launch_bounds__(256) void vq_main(const float* __restrict__ lat,
                                               const float* __restrict__ emb,
                                               const _Float16* __restrict__ ehi,
                                               const _Float16* __restrict__ elo,
                                               const float* __restrict__ cnorm,
                                               float* __restrict__ out,
                                               float* __restrict__ partial) {
    __shared__ int   sIdx[ROWS_PER_BLOCK];
    __shared__ float sRed[256];

    const int tid  = threadIdx.x;
    const int wave = tid >> 5;
    const int lane = tid & 31;
    const int hl   = lane >> 4;   // lane half: selects K-range of A / B
    const int c    = lane & 15;   // A row (M) index / C-D column (code) index

    const int nbase  = blockIdx.x * ROWS_PER_BLOCK;   // flat (b,h,w) row base
    const int bImg   = nbase >> 10;                   // batch index
    const int hwbase = nbase & 1023;                  // block never crosses a batch image

    // ---- Load two A row-tiles (rows n0+0..15 and n0+16..31) ----
    const float* latb0 = lat + bImg * (DIM * 1024) + hwbase + wave * ROWS_PER_WAVE + c;
    const float* latb1 = latb0 + 16;

    v16h aH0[2], aL0[2], aH1[2], aL1[2];
    load_split_A(latb0, hl, aH0, aL0);
    load_split_A(latb1, hl, aH1, aL1);

    float bv0[8], bv1[8];
    int   bi0[8], bi1[8];
    #pragma unroll
    for (int r = 0; r < 8; ++r) {
        bv0[r] = 3.4e38f; bv1[r] = 3.4e38f;
        bi0[r] = 0;       bi1[r] = 0;
    }

    // ---- Sweep all 512 codes in 16-code tiles; both row-tiles share B ----
    for (int t = 0; t < KCODES / 16; ++t) {
        const int code = t * 16 + c;
        // B 32x16 in this code's k order (matches A): lane column n=c, VGPR = consecutive dim pair,
        // lane half hl selects K = hl*16 + 0..15 -> contiguous 32B of [code][dim].
        const _Float16* eh = ehi + code * DIM + hl * 16;
        const _Float16* el = elo + code * DIM + hl * 16;
        v8f acc0 = {}, acc1 = {};
        #pragma unroll
        for (int s = 0; s < 2; ++s) {
            v16h bh = *(const v16h*)(eh + s * 32);
            v16h bl = *(const v16h*)(el + s * 32);
            // compensated product: x_hi*e_hi + x_lo*e_hi + x_hi*e_lo,
            // interleaved across the two independent accumulator chains
            acc0 = WMMA_F16(aH0[s], bh, acc0);
            acc1 = WMMA_F16(aH1[s], bh, acc1);
            acc0 = WMMA_F16(aL0[s], bh, acc0);
            acc1 = WMMA_F16(aL1[s], bh, acc1);
            acc0 = WMMA_F16(aH0[s], bl, acc0);
            acc1 = WMMA_F16(aH1[s], bl, acc1);
        }
        const float cn = cnorm[code];
        // dist (minus constant ||x||^2) = ||e||^2 - 2 x.e ; row-in-tile = hl*8 + r
        #pragma unroll
        for (int r = 0; r < 8; ++r) {
            float d0 = fmaf(-2.0f, acc0[r], cn);
            float d1 = fmaf(-2.0f, acc1[r], cn);
            if (d0 < bv0[r]) { bv0[r] = d0; bi0[r] = code; }  // ascending code =>
            if (d1 < bv1[r]) { bv1[r] = d1; bi1[r] = code; }  // first-min kept
        }
    }

    // ---- Butterfly argmin across the 16 lanes sharing the same rows ----
    #pragma unroll
    for (int m = 8; m >= 1; m >>= 1) {
        #pragma unroll
        for (int r = 0; r < 8; ++r) {
            float ov0 = __shfl_xor(bv0[r], m, 32);
            int   oi0 = __shfl_xor(bi0[r], m, 32);
            if (ov0 < bv0[r] || (ov0 == bv0[r] && oi0 < bi0[r])) { bv0[r] = ov0; bi0[r] = oi0; }
            float ov1 = __shfl_xor(bv1[r], m, 32);
            int   oi1 = __shfl_xor(bi1[r], m, 32);
            if (ov1 < bv1[r] || (ov1 == bv1[r] && oi1 < bi1[r])) { bv1[r] = ov1; bi1[r] = oi1; }
        }
    }
    if (c == 0) {
        #pragma unroll
        for (int r = 0; r < 8; ++r) {
            sIdx[wave * ROWS_PER_WAVE + hl * 8 + r]      = bi0[r];
            sIdx[wave * ROWS_PER_WAVE + 16 + hl * 8 + r] = bi1[r];
        }
    }
    __syncthreads();

    // ---- Gather exact fp32 codes, write output, accumulate loss ----
    float lsum = 0.f;
    const float* latB = lat + bImg * (DIM * 1024) + hwbase;
    float*       outB = out + bImg * (DIM * 1024) + hwbase;
    for (int idx = tid; idx < ROWS_PER_BLOCK * DIM; idx += 256) {
        int d = idx >> 8;                        // 0..63
        int r = idx & 255;                       // 0..255, contiguous per wave
        float ev = emb[sIdx[r] * DIM + d];       // exact reference value
        float xv = latB[d * 1024 + r];
        outB[d * 1024 + r] = ev;                 // coalesced write
        float df = ev - xv;
        lsum = fmaf(df, df, lsum);
    }
    sRed[tid] = lsum;
    __syncthreads();
    #pragma unroll
    for (int sft = 128; sft > 0; sft >>= 1) {
        if (tid < sft) sRed[tid] += sRed[tid + sft];
        __syncthreads();
    }
    if (tid == 0) partial[blockIdx.x] = sRed[0];
}

// ---------------------------------------------------------------------------
// Final deterministic loss reduction: 256 partials -> scalar.
// ---------------------------------------------------------------------------
__global__ __launch_bounds__(256) void vq_loss_reduce(const float* __restrict__ partial,
                                                      float* __restrict__ lossOut) {
    __shared__ float s[256];
    int t = threadIdx.x;
    s[t] = partial[t];
    __syncthreads();
    #pragma unroll
    for (int sft = 128; sft > 0; sft >>= 1) {
        if (t < sft) s[t] += s[t + sft];
        __syncthreads();
    }
    // vq_loss = beta*mean + mean = 1.25 * mean over N*D elements
    if (t == 0) lossOut[0] = s[0] * (1.25f / (float)(NROWS * DIM));
}

extern "C" void kernel_launch(void* const* d_in, const int* in_sizes, int n_in,
                              void* d_out, int out_size, void* d_ws, size_t ws_size,
                              hipStream_t stream) {
    const float* latents = (const float*)d_in[0];   // [64, 64, 32, 32] f32
    const float* emb     = (const float*)d_in[1];   // [512, 64] f32
    float* out = (float*)d_out;                     // 64*64*32*32 quant + 1 loss

    // workspace layout
    _Float16* ehi  = (_Float16*)d_ws;               // 512*64 f16  (64 KB)
    _Float16* elo  = ehi + KCODES * DIM;            // 512*64 f16  (64 KB)
    float* cnorm   = (float*)(elo + KCODES * DIM);  // 512 f32
    float* partial = cnorm + KCODES;                // 256 f32 (one per block)

    vq_prep<<<2, 256, 0, stream>>>(emb, ehi, elo, cnorm);
    vq_main<<<NROWS / ROWS_PER_BLOCK, 256, 0, stream>>>(latents, emb, ehi, elo,
                                                        cnorm, out, partial);
    vq_loss_reduce<<<1, 256, 0, stream>>>(partial, out + (size_t)NROWS * DIM);
}
